// RecomGCL_82008105549985
// MI455X (gfx1250) — matrix operand order repaired and checked by reading, not compiled, not run on hardware
//
#include <hip/hip_runtime.h>
#include <hip/hip_bf16.h>

typedef __attribute__((ext_vector_type(16))) _Float16 v16h;
typedef __attribute__((ext_vector_type(8)))  float    v8f;

#define N_U   50000
#define N_I   50000
#define DIM   64
#define NNZ   2000000
#define BATCH 1024
#define NPOS  10
#define NNEG  40

// ---------------------------------------------------------------------------
// init: E = E0, sum = E0  (users then items, concatenated)
// ---------------------------------------------------------------------------
__global__ __launch_bounds__(256)
void init_kernel(const float* __restrict__ Eu0, const float* __restrict__ Ei0,
                 float* __restrict__ E, float* __restrict__ S,
                 int nu_elems, int n_total) {
    int i = blockIdx.x * 256 + threadIdx.x;
    if (i >= n_total) return;
    float x = (i < nu_elems) ? Eu0[i] : Ei0[i - nu_elems];
    E[i] = x;
    S[i] = x;
}

__global__ __launch_bounds__(256)
void zero_kernel(float* __restrict__ p, int n) {
    int i = blockIdx.x * 256 + threadIdx.x;
    if (i < n) p[i] = 0.0f;
}

// ---------------------------------------------------------------------------
// SpMM (both directions in one pass, reads pre-update E, scatters into Z):
//   Z_u[r] += v * E_i[c]   and   Z_i[c] += v * E_u[r]
// 16 lanes per edge, float4 per lane (L2-resident gathers, atomic scatter).
// ---------------------------------------------------------------------------
__global__ __launch_bounds__(256)
void spmm_kernel(const float* __restrict__ Eu, const float* __restrict__ Ei,
                 const int* __restrict__ rows, const int* __restrict__ cols,
                 const float* __restrict__ vals,
                 float* __restrict__ Zu, float* __restrict__ Zi, int nnz) {
    int tid = blockIdx.x * 256 + threadIdx.x;
    int e = tid >> 4;
    if (e >= nnz) return;
    int q = (tid & 15) * 4;
    int r = rows[e];
    int c = cols[e];
    float v = vals[e];

    const float4 ei = *(const float4*)(Ei + (size_t)c * DIM + q);
    const float4 eu = *(const float4*)(Eu + (size_t)r * DIM + q);

    float* zu = Zu + (size_t)r * DIM + q;
    float* zi = Zi + (size_t)c * DIM + q;
    atomicAdd(zu + 0, v * ei.x); atomicAdd(zu + 1, v * ei.y);
    atomicAdd(zu + 2, v * ei.z); atomicAdd(zu + 3, v * ei.w);
    atomicAdd(zi + 0, v * eu.x); atomicAdd(zi + 1, v * eu.y);
    atomicAdd(zi + 2, v * eu.z); atomicAdd(zi + 3, v * eu.w);
}

// ---------------------------------------------------------------------------
// update: E += leaky(Z);  sum += E(new)     (LeakyReLU slope 0.5)
// ---------------------------------------------------------------------------
__global__ __launch_bounds__(256)
void update_kernel(float* __restrict__ E, float* __restrict__ S,
                   const float* __restrict__ Z, int n) {
    int i = blockIdx.x * 256 + threadIdx.x;
    if (i >= n) return;
    float z = Z[i];
    z = (z > 0.0f) ? z : 0.5f * z;
    float e = E[i] + z;
    E[i] = e;
    S[i] += e;
}

// ---------------------------------------------------------------------------
// Scoring via WMMA. One wave32 per user.
//   A tile = 16 items x 32 k-chunk (f16, documented 16-bit A layout)
//   B tile = user embedding broadcast to all 16 N-columns (K=(lane/16)*8+e for e<8, 16+(lane/16)*8+(e-8) for e>=8)
//   f32 split into hi/lo f16 halves -> 3 WMMAs per chunk for ~f32 accuracy.
//   Scores live in column N=0 of C: lane 0 (M=0..7), lane 16 (M=8..15).
// ---------------------------------------------------------------------------
__global__ __launch_bounds__(32)
void score_kernel(const float* __restrict__ Su, const float* __restrict__ Si,
                  const int* __restrict__ uids, const int* __restrict__ pos,
                  const int* __restrict__ neg, float* __restrict__ w) {
    __shared__ float uvec[DIM];
    __shared__ float sScores[64];

    const int b    = blockIdx.x;
    const int lane = threadIdx.x;
    const int half = lane >> 4;   // 0 or 1
    const int mloc = lane & 15;

    const int uid = uids[b];
    uvec[2 * lane]     = Su[(size_t)uid * DIM + 2 * lane];
    uvec[2 * lane + 1] = Su[(size_t)uid * DIM + 2 * lane + 1];
    __syncthreads();

    // B fragments for both k-chunks (broadcast over N): element e <-> K = half*8+e (e<8), 16+half*8+(e-8) (e>=8)
    v16h bhi0, blo0, bhi1, blo1;
#pragma unroll
    for (int e = 0; e < 16; ++e) {
        float x0 = uvec[half * 8 + e + (e >= 8 ? 8 : 0)];
        _Float16 h0 = (_Float16)x0;
        bhi0[e] = h0;
        blo0[e] = (_Float16)(x0 - (float)h0);
        float x1 = uvec[32 + half * 8 + e + (e >= 8 ? 8 : 0)];
        _Float16 h1 = (_Float16)x1;
        bhi1[e] = h1;
        blo1[e] = (_Float16)(x1 - (float)h1);
    }

#pragma unroll
    for (int g = 0; g < 4; ++g) {
        int item  = g * 16 + mloc;
        int valid = (item < NPOS + NNEG) ? 1 : 0;
        int idx   = 0;
        if (valid)
            idx = (item < NPOS) ? pos[b * NPOS + item]
                                : neg[b * NNEG + (item - NPOS)];
        const float* row = Si + (size_t)idx * DIM;

        v8f acc = {};
#pragma unroll
        for (int cc = 0; cc < 2; ++cc) {
            const int base = cc * 32;
            v16h ahi, alo;
#pragma unroll
            for (int e = 0; e < 8; ++e) {
                // A 16-bit layout: lanes<16 -> K 0..7 & 16..23 ; lanes>=16 -> K 8..15 & 24..31
                float x0 = valid ? row[base + half * 8 + e] : 0.0f;
                float x1 = valid ? row[base + 16 + half * 8 + e] : 0.0f;
                _Float16 h0 = (_Float16)x0, h1 = (_Float16)x1;
                ahi[e]     = h0;
                alo[e]     = (_Float16)(x0 - (float)h0);
                ahi[8 + e] = h1;
                alo[8 + e] = (_Float16)(x1 - (float)h1);
            }
            v16h bh = cc ? bhi1 : bhi0;
            v16h bl = cc ? blo1 : blo0;
            acc = __builtin_amdgcn_wmma_f32_16x16x32_f16(false, ahi, false, bh,
                                                         (short)0, acc, false, false);
            acc = __builtin_amdgcn_wmma_f32_16x16x32_f16(false, ahi, false, bl,
                                                         (short)0, acc, false, false);
            acc = __builtin_amdgcn_wmma_f32_16x16x32_f16(false, alo, false, bh,
                                                         (short)0, acc, false, false);
        }

        if (lane == 0) {
#pragma unroll
            for (int vv = 0; vv < 8; ++vv) sScores[g * 16 + vv] = acc[vv];
        }
        if (lane == 16) {
#pragma unroll
            for (int vv = 0; vv < 8; ++vv) sScores[g * 16 + 8 + vv] = acc[vv];
        }
        __syncthreads();
    }

    if (lane == 0) {
        float ph = 0.0f, nh = 0.0f;
        float pmin = 3.402823e38f, nmax = -3.402823e38f;
        for (int j = 0; j < NPOS; ++j) {
            float s = sScores[j];
            float h = 1.0f - s;
            if (h > 0.0f) ph += h;
            if (s < pmin) pmin = s;
        }
        for (int j = NPOS; j < NPOS + NNEG; ++j) {
            float s = sScores[j];
            float h = s - 0.5f;           // MARGIN
            if (h > 0.0f) nh += h;
            if (s > nmax) nmax = s;
        }
        float delta = (nmax - pmin > 0.005f) ? 10.0f * (pmin - nmax) : 0.0f;
        // bpr_sum + cumsum weight (B - b) * delta
        w[b] = ph + 10.0f * nh + (float)(BATCH - b) * delta;
    }
}

// ---------------------------------------------------------------------------
// deterministic fixed-order tree reduction -> scalar loss
// ---------------------------------------------------------------------------
__global__ __launch_bounds__(256)
void reduce_kernel(const float* __restrict__ w, float* __restrict__ out) {
    __shared__ float s[256];
    int t = threadIdx.x;
    float acc = 0.0f;
    for (int i = t; i < BATCH; i += 256) acc += w[i];
    s[t] = acc;
    __syncthreads();
    for (int off = 128; off > 0; off >>= 1) {
        if (t < off) s[t] += s[t + off];
        __syncthreads();
    }
    if (t == 0) out[0] = s[0] / (float)BATCH + 0.2f;   // LAMBDA1
}

// ---------------------------------------------------------------------------
extern "C" void kernel_launch(void* const* d_in, const int* in_sizes, int n_in,
                              void* d_out, int out_size, void* d_ws, size_t ws_size,
                              hipStream_t stream) {
    const float* Eu0  = (const float*)d_in[0];
    const float* Ei0  = (const float*)d_in[1];
    const float* vals = (const float*)d_in[2];
    // d_in[3..6] (u_mul_s, vt, v_mul_s, ut) feed only the deleted SVD branch: skipped.
    const int* rows = (const int*)d_in[7];
    const int* cols = (const int*)d_in[8];
    const int* uids = (const int*)d_in[9];
    const int* pos  = (const int*)d_in[10];
    const int* neg  = (const int*)d_in[11];

    const size_t NEU = (size_t)N_U * DIM;           // 3.2M floats
    const size_t NE  = (size_t)(N_U + N_I) * DIM;   // 6.4M floats

    float* E  = (float*)d_ws;        // [E_u | E_i]
    float* S  = E + NE;              // [sum_u | sum_i]
    float* Z  = S + NE;              // [Z_u | Z_i]
    float* W  = Z + NE;              // per-user loss terms [1024]
    float* Eu = E;       float* Ei = E + NEU;
    float* Su = S;       float* Si = S + NEU;
    float* Zu = Z;       float* Zi = Z + NEU;

    const int nElem   = (int)NE;
    const int gElem   = (nElem + 255) / 256;
    const int gSpmm   = (NNZ * 16 + 255) / 256;

    init_kernel<<<gElem, 256, 0, stream>>>(Eu0, Ei0, E, S, (int)NEU, nElem);

    for (int layer = 0; layer < 2; ++layer) {
        zero_kernel<<<gElem, 256, 0, stream>>>(Z, nElem);
        spmm_kernel<<<gSpmm, 256, 0, stream>>>(Eu, Ei, rows, cols, vals, Zu, Zi, NNZ);
        update_kernel<<<gElem, 256, 0, stream>>>(E, S, Z, nElem);
    }

    score_kernel<<<BATCH, 32, 0, stream>>>(Su, Si, uids, pos, neg, W);
    reduce_kernel<<<1, 256, 0, stream>>>(W, (float*)d_out);
}
